// Attention_43808666419840
// MI455X (gfx1250) — matrix operand order repaired and checked by reading, not compiled, not run on hardware
//
#include <hip/hip_runtime.h>
#include <hip/hip_bf16.h>

typedef __attribute__((ext_vector_type(16))) __bf16 v16bf;
typedef __attribute__((ext_vector_type(8)))  __bf16 v8bf;
typedef __attribute__((ext_vector_type(8)))  float  v8f;

#if defined(__has_builtin)
#if __has_builtin(__builtin_amdgcn_sched_barrier)
#define SCHED_FENCE() __builtin_amdgcn_sched_barrier(0)
#endif
#endif
#ifndef SCHED_FENCE
#define SCHED_FENCE() ((void)0)
#endif

namespace {
constexpr int kHeads = 12;
constexpr int kHd    = 64;
constexpr int kHdim  = 32;               // spatial H
constexpr int kWdim  = 32;               // spatial W
constexpr int kSeq   = kHdim * kWdim;    // 1024
constexpr int kB     = 4;
constexpr int kBH    = kB * kHeads;      // 48
constexpr int kC     = 768;
constexpr int kM     = kB * kSeq;        // 4096 rows of x
constexpr int kNqkv  = 3 * kC;           // 2304
constexpr int kLd    = 80;               // padded LDS leading dim (bf16), 160B rows
}

// ---------- small helpers ----------

__device__ __forceinline__ __bf16 f2bf(float x) { return (__bf16)x; }
__device__ __forceinline__ float  bf2f(__bf16 x) { return (float)x; }

__device__ __forceinline__ v8f wmma_bf16(v16bf a, v16bf b, v8f c) {
  return __builtin_amdgcn_wmma_f32_16x16x32_bf16(
      /*neg_a=*/false, a, /*neg_b=*/false, b,
      /*c_mod=*/(short)0, c, /*reuse_a=*/false, /*reuse_b=*/false);
}

__device__ __forceinline__ v16bf cat16(v8bf lo, v8bf hi) {
  return __builtin_shufflevector(lo, hi, 0, 1, 2, 3, 4, 5, 6, 7,
                                 8, 9, 10, 11, 12, 13, 14, 15);
}

// A fragment (16x32 bf16, MxK) from row-major [16][ld] LDS tile.
// lane: row = lane&15, half = lane>>4. K runs: [8h..8h+7] and [16+8h..16+8h+7]
// -> two contiguous 16B ds_load_b128.
__device__ __forceinline__ v16bf load_a_frag(const __bf16* p, int ld) {
  const int lane = threadIdx.x & 31;
  const __bf16* base = p + (lane & 15) * ld + ((lane >> 4) << 3);
  const v8bf lo = *(const v8bf*)(base);
  const v8bf hi = *(const v8bf*)(base + 16);
  return cat16(lo, hi);
}

// B fragment (KxN logical, K=32,N=16) from an LDS tile stored [N][K] row-major
// (K contiguous per column). lane: n = lane&15, half = lane>>4. K runs: [8h..8h+7] and [16+8h..16+8h+7]
// -> two contiguous 16B ds_load_b128.
__device__ __forceinline__ v16bf load_b_frag_nk(const __bf16* p, int ld) {
  const int lane = threadIdx.x & 31;
  const __bf16* base = p + (lane & 15) * ld + ((lane >> 4) << 3);
  const v8bf lo = *(const v8bf*)(base);
  const v8bf hi = *(const v8bf*)(base + 16);
  return cat16(lo, hi);
}

// ---------- kernel 1: qkv = x @ Wqkv, scattered to head-major bf16 ----------
// Block: 128 threads (4 waves). Output tile 128(M) x 64(N); wave owns 32 rows.

__global__ __launch_bounds__(128)
void qkv_gemm(const float* __restrict__ x, const float* __restrict__ w,
              __bf16* __restrict__ qo, __bf16* __restrict__ ko,
              __bf16* __restrict__ vo) {
  __shared__ __bf16 As[128 * kLd];   // [M][K]
  __shared__ __bf16 Bs[64 * kLd];    // [N][K] (transposed at fill)
  const int tid  = threadIdx.x;
  const int wave = tid >> 5;
  const int lane = tid & 31;
  const int half = lane >> 4;
  const int ncol = lane & 15;
  const int m0 = blockIdx.x * 128;
  const int n0 = blockIdx.y * 64;

  v8f acc[2][4] = {};
  for (int kk = 0; kk < kC; kk += 64) {
    __syncthreads();
    for (int i = tid * 4; i < 128 * 64; i += 128 * 4) {
      const int r = i >> 6, c = i & 63;
      const float4 f = *(const float4*)(x + (size_t)(m0 + r) * kC + kk + c);
      __bf16* d = &As[r * kLd + c];
      d[0] = f2bf(f.x); d[1] = f2bf(f.y); d[2] = f2bf(f.z); d[3] = f2bf(f.w);
    }
    for (int i = tid * 4; i < 64 * 64; i += 128 * 4) {
      const int r = i >> 6, c = i & 63;   // r: K row of W, c: N col
      const float4 g = *(const float4*)(w + (size_t)(kk + r) * kNqkv + n0 + c);
      Bs[(c + 0) * kLd + r] = f2bf(g.x);
      Bs[(c + 1) * kLd + r] = f2bf(g.y);
      Bs[(c + 2) * kLd + r] = f2bf(g.z);
      Bs[(c + 3) * kLd + r] = f2bf(g.w);
    }
    __syncthreads();
    // warm L2/WGP$ for the next K-chunk while this one computes
    if (kk + 64 < kC) {
      __builtin_prefetch(x + (size_t)(m0 + tid) * kC + kk + 64, 0, 1);
      __builtin_prefetch(w + (size_t)(kk + 64 + (tid >> 1)) * kNqkv + n0, 0, 1);
    }
    // Batch ALL fragment loads, then fence the scheduler so they are not
    // re-sunk into the WMMA sequence (keeps the matrix pipe fed).
    v16bf a0[2], a1[2], b0[4], b1[4];
#pragma unroll
    for (int s = 0; s < 2; ++s) {
      a0[s] = load_a_frag(&As[(wave * 32 + s * 16) * kLd + 0],  kLd);
      a1[s] = load_a_frag(&As[(wave * 32 + s * 16) * kLd + 32], kLd);
    }
#pragma unroll
    for (int nt = 0; nt < 4; ++nt) {
      b0[nt] = load_b_frag_nk(&Bs[(nt * 16) * kLd + 0],  kLd);
      b1[nt] = load_b_frag_nk(&Bs[(nt * 16) * kLd + 32], kLd);
    }
    SCHED_FENCE();
#pragma unroll
    for (int nt = 0; nt < 4; ++nt)
#pragma unroll
      for (int s = 0; s < 2; ++s) {
        acc[s][nt] = wmma_bf16(a0[s], b0[nt], acc[s][nt]);
        acc[s][nt] = wmma_bf16(a1[s], b1[nt], acc[s][nt]);
      }
    SCHED_FENCE();
  }

  // scatter into q/k/v (48, 1024, 64) bf16; n0 block lies inside one of q/k/v
  const int which = n0 / kC;                    // uniform per block
  __bf16* dst = (which == 0) ? qo : (which == 1) ? ko : vo;
#pragma unroll
  for (int nt = 0; nt < 4; ++nt) {
    const int n    = n0 - which * kC + nt * 16 + ncol;  // 0..767 within q/k/v
    const int head = n >> 6, dch = n & 63;
#pragma unroll
    for (int s = 0; s < 2; ++s)
#pragma unroll
      for (int r = 0; r < 8; ++r) {
        const int m = m0 + wave * 32 + s * 16 + half * 8 + r;
        const int b = m >> 10, sq = m & 1023;
        dst[((size_t)(b * kHeads + head) * kSeq + sq) * kHd + dch] =
            f2bf(acc[s][nt][r]);
      }
  }
}

// ---------- kernel 2: decomposed relative-position bias ----------

__global__ __launch_bounds__(256)
void relbias(const __bf16* __restrict__ q, const float* __restrict__ rph,
             const float* __restrict__ rpw, float* __restrict__ rel_h,
             float* __restrict__ rel_w) {
  const int idx = blockIdx.x * blockDim.x + threadIdx.x;  // over 48*1024*32
  const int j  = idx & 31;
  const int bq = idx >> 5;
  const int s  = bq & 1023;
  const int qh = s >> 5, qw = s & 31;
  const __bf16* qp = q + (size_t)bq * kHd;
  const float* ph = rph + (size_t)(qh - j + 31) * kHd;
  const float* pw = rpw + (size_t)(qw - j + 31) * kHd;
  float ah = 0.f, aw = 0.f;
#pragma unroll 8
  for (int c = 0; c < kHd; ++c) {
    const float qc = bf2f(qp[c]);
    ah += qc * ph[c];
    aw += qc * pw[c];
  }
  rel_h[idx] = ah;
  rel_w[idx] = aw;
}

// ---------- kernel 3: fused flash attention with additive bias ----------
// One block = one (head, 64-query tile). 4 waves; wave owns 16 query rows.

__global__ __launch_bounds__(128)
void attn_fused(const __bf16* __restrict__ q, const __bf16* __restrict__ k,
                const __bf16* __restrict__ v, const float* __restrict__ rel_h,
                const float* __restrict__ rel_w, __bf16* __restrict__ ctx) {
  __shared__ __bf16 Qs[64 * kLd];        // [query][hd]
  __shared__ __bf16 Ks[64 * kLd];        // [key][hd]   == [N][K] for q@k^T
  __shared__ __bf16 VsT[64 * kLd];       // [hd][key]   == [N][K] for P@V
  __shared__ __bf16 Ps[4 * 16 * kLd];    // per-wave P strip [row][K]
  __shared__ float  Rh[64 * kHdim];
  __shared__ float  Rw[64 * kWdim];

  const int tid  = threadIdx.x;
  const int wave = tid >> 5;
  const int lane = tid & 31;
  const int half = lane >> 4;
  const int ncol = lane & 15;
  const int bh = blockIdx.y;
  const int q0 = blockIdx.x * 64;
  const size_t headoff = (size_t)bh * kSeq * kHd;

  // Q tile, pre-scaled by hd^-0.5 = 0.125 (exact power-of-two in bf16)
  for (int i = tid; i < 64 * 64; i += 128) {
    const int r = i >> 6, c = i & 63;
    Qs[r * kLd + c] = f2bf(bf2f(q[headoff + (size_t)(q0 + r) * kHd + c]) * 0.125f);
  }
  // bias rows for these 64 queries
  for (int i = tid * 4; i < 64 * kHdim; i += 128 * 4) {
    const int r = i >> 5, c = i & 31;
    *(float4*)&Rh[r * kHdim + c] =
        *(const float4*)&rel_h[((size_t)bh * kSeq + q0 + r) * kHdim + c];
    *(float4*)&Rw[r * kWdim + c] =
        *(const float4*)&rel_w[((size_t)bh * kSeq + q0 + r) * kWdim + c];
  }
  __syncthreads();

  const v16bf qa0 = load_a_frag(&Qs[(wave * 16) * kLd + 0],  kLd);
  const v16bf qa1 = load_a_frag(&Qs[(wave * 16) * kLd + 32], kLd);

  v8f acc[4] = {};
  float m_i[8], l_i[8];
#pragma unroll
  for (int r = 0; r < 8; ++r) { m_i[r] = -1e30f; l_i[r] = 0.f; }
  __bf16* Pw = &Ps[(wave * 16) * kLd];
  const int rowbase = wave * 16 + half * 8;

  for (int kt = 0; kt < kSeq / 64; ++kt) {
    __syncthreads();
    for (int i = tid * 8; i < 64 * 64; i += 128 * 8) {
      const int r = i >> 6, c = i & 63;     // r: key, c: hd
      *(uint4*)&Ks[r * kLd + c] =
          *(const uint4*)&k[headoff + (size_t)(kt * 64 + r) * kHd + c];
      const uint4 raw =
          *(const uint4*)&v[headoff + (size_t)(kt * 64 + r) * kHd + c];
      const __bf16* e = (const __bf16*)&raw;
#pragma unroll
      for (int j = 0; j < 8; ++j) VsT[(c + j) * kLd + r] = e[j];
    }
    __syncthreads();
    // prefetch next K/V tiles into cache while we compute on this one
    if (kt + 1 < kSeq / 64) {
      const size_t nxt = headoff + (size_t)((kt + 1) * 64 + (tid >> 1)) * kHd;
      __builtin_prefetch(&k[nxt], 0, 1);
      __builtin_prefetch(&v[nxt], 0, 1);
    }

    // scores: 16x64 per wave = 4 WMMA C tiles, K=64 (2 WMMAs each).
    // Batch K-fragments and fence so they stay ahead of the WMMA burst.
    v16bf kb0[4], kb1[4];
#pragma unroll
    for (int nt = 0; nt < 4; ++nt) {
      kb0[nt] = load_b_frag_nk(&Ks[(nt * 16) * kLd + 0],  kLd);
      kb1[nt] = load_b_frag_nk(&Ks[(nt * 16) * kLd + 32], kLd);
    }
    SCHED_FENCE();
    v8f s[4];
#pragma unroll
    for (int nt = 0; nt < 4; ++nt) {
      v8f sc = {};
      sc = wmma_bf16(qa0, kb0[nt], sc);
      sc = wmma_bf16(qa1, kb1[nt], sc);
      const int col = kt * 64 + nt * 16 + ncol;
      const int kh = col >> 5, kw = col & 31;
#pragma unroll
      for (int r = 0; r < 8; ++r)
        sc[r] += Rh[(rowbase + r) * kHdim + kh] + Rw[(rowbase + r) * kWdim + kw];
      s[nt] = sc;
    }

    // V fragments issued now: their ds_load latency hides under softmax VALU.
    v16bf vb0[4], vb1[4];
#pragma unroll
    for (int nt = 0; nt < 4; ++nt) {
      vb0[nt] = load_b_frag_nk(&VsT[(nt * 16) * kLd + 0],  kLd);
      vb1[nt] = load_b_frag_nk(&VsT[(nt * 16) * kLd + 32], kLd);
    }
    SCHED_FENCE();

    // online softmax: rows live in 16-lane halves -> xor reductions 1,2,4,8
    float rmax[8];
#pragma unroll
    for (int r = 0; r < 8; ++r) {
      rmax[r] = fmaxf(fmaxf(s[0][r], s[1][r]), fmaxf(s[2][r], s[3][r]));
      rmax[r] = fmaxf(rmax[r], __shfl_xor(rmax[r], 1, 32));
      rmax[r] = fmaxf(rmax[r], __shfl_xor(rmax[r], 2, 32));
      rmax[r] = fmaxf(rmax[r], __shfl_xor(rmax[r], 4, 32));
      rmax[r] = fmaxf(rmax[r], __shfl_xor(rmax[r], 8, 32));
    }
    float alpha[8], rsum[8];
#pragma unroll
    for (int r = 0; r < 8; ++r) {
      const float mn = fmaxf(m_i[r], rmax[r]);
      alpha[r] = __expf(m_i[r] - mn);
      m_i[r] = mn;
      rsum[r] = 0.f;
    }
#pragma unroll
    for (int nt = 0; nt < 4; ++nt)
#pragma unroll
      for (int r = 0; r < 8; ++r) {
        const float p = __expf(s[nt][r] - m_i[r]);
        s[nt][r] = p;
        rsum[r] += p;
      }
#pragma unroll
    for (int r = 0; r < 8; ++r) {
      rsum[r] += __shfl_xor(rsum[r], 1, 32);
      rsum[r] += __shfl_xor(rsum[r], 2, 32);
      rsum[r] += __shfl_xor(rsum[r], 4, 32);
      rsum[r] += __shfl_xor(rsum[r], 8, 32);
      l_i[r] = l_i[r] * alpha[r] + rsum[r];
    }
    // rescale O and stage P (bf16) in this wave's private LDS strip
#pragma unroll
    for (int nt = 0; nt < 4; ++nt)
#pragma unroll
      for (int r = 0; r < 8; ++r) {
        acc[nt][r] *= alpha[r];
        Pw[(half * 8 + r) * kLd + nt * 16 + ncol] = f2bf(s[nt][r]);
      }
    // P @ V  (same-wave LDS write->read is in order; no barrier needed)
    const v16bf pa0 = load_a_frag(&Pw[0],  kLd);
    const v16bf pa1 = load_a_frag(&Pw[32], kLd);
#pragma unroll
    for (int nt = 0; nt < 4; ++nt) {
      acc[nt] = wmma_bf16(pa0, vb0[nt], acc[nt]);
      acc[nt] = wmma_bf16(pa1, vb1[nt], acc[nt]);
    }
  }

  // normalize; write ctx in (B, S, C) layout so proj GEMM reads it row-major
  const int b = bh / kHeads, head = bh % kHeads;
  float inv[8];
#pragma unroll
  for (int r = 0; r < 8; ++r) inv[r] = 1.f / l_i[r];
#pragma unroll
  for (int nt = 0; nt < 4; ++nt)
#pragma unroll
    for (int r = 0; r < 8; ++r) {
      const int sidx = q0 + wave * 16 + half * 8 + r;
      const int colc = head * kHd + nt * 16 + ncol;
      ctx[((size_t)(b * kSeq) + sidx) * kC + colc] = f2bf(acc[nt][r] * inv[r]);
    }
}

// ---------- kernel 4: out = ctx @ Wproj + bproj ----------
// Block: 128 threads; output tile 128(M) x 64(N); wave owns 32 rows.

__global__ __launch_bounds__(128)
void proj_gemm(const __bf16* __restrict__ ctx, const float* __restrict__ w,
               const float* __restrict__ bias, float* __restrict__ out) {
  __shared__ __bf16 As[128 * kLd];   // [M][K]
  __shared__ __bf16 Bs[64 * kLd];    // [N][K]
  const int tid  = threadIdx.x;
  const int wave = tid >> 5;
  const int lane = tid & 31;
  const int half = lane >> 4;
  const int ncol = lane & 15;
  const int m0 = blockIdx.x * 128;
  const int n0 = blockIdx.y * 64;

  v8f acc[2][4] = {};
  for (int kk = 0; kk < kC; kk += 64) {
    __syncthreads();
    for (int i = tid * 8; i < 128 * 64; i += 128 * 8) {
      const int r = i >> 6, c = i & 63;
      *(uint4*)&As[r * kLd + c] =
          *(const uint4*)&ctx[(size_t)(m0 + r) * kC + kk + c];
    }
    for (int i = tid * 4; i < 64 * 64; i += 128 * 4) {
      const int r = i >> 6, c = i & 63;
      const float4 g = *(const float4*)(w + (size_t)(kk + r) * kC + n0 + c);
      Bs[(c + 0) * kLd + r] = f2bf(g.x);
      Bs[(c + 1) * kLd + r] = f2bf(g.y);
      Bs[(c + 2) * kLd + r] = f2bf(g.z);
      Bs[(c + 3) * kLd + r] = f2bf(g.w);
    }
    __syncthreads();
    if (kk + 64 < kC) {
      __builtin_prefetch(ctx + (size_t)(m0 + tid) * kC + kk + 64, 0, 1);
      __builtin_prefetch(w + (size_t)(kk + 64 + (tid >> 1)) * kC + n0, 0, 1);
    }
    v16bf a0[2], a1[2], b0[4], b1[4];
#pragma unroll
    for (int s = 0; s < 2; ++s) {
      a0[s] = load_a_frag(&As[(wave * 32 + s * 16) * kLd + 0],  kLd);
      a1[s] = load_a_frag(&As[(wave * 32 + s * 16) * kLd + 32], kLd);
    }
#pragma unroll
    for (int nt = 0; nt < 4; ++nt) {
      b0[nt] = load_b_frag_nk(&Bs[(nt * 16) * kLd + 0],  kLd);
      b1[nt] = load_b_frag_nk(&Bs[(nt * 16) * kLd + 32], kLd);
    }
    SCHED_FENCE();
#pragma unroll
    for (int nt = 0; nt < 4; ++nt)
#pragma unroll
      for (int s = 0; s < 2; ++s) {
        acc[s][nt] = wmma_bf16(a0[s], b0[nt], acc[s][nt]);
        acc[s][nt] = wmma_bf16(a1[s], b1[nt], acc[s][nt]);
      }
    SCHED_FENCE();
  }
#pragma unroll
  for (int nt = 0; nt < 4; ++nt) {
    const int n = n0 + nt * 16 + ncol;
    const float bn = bias[n];
#pragma unroll
    for (int s = 0; s < 2; ++s)
#pragma unroll
      for (int r = 0; r < 8; ++r) {
        const int m = m0 + wave * 32 + s * 16 + half * 8 + r;
        out[(size_t)m * kC + n] = acc[s][nt][r] + bn;
      }
  }
}

// ---------- launcher ----------

extern "C" void kernel_launch(void* const* d_in, const int* in_sizes, int n_in,
                              void* d_out, int out_size, void* d_ws, size_t ws_size,
                              hipStream_t stream) {
  const float* x     = (const float*)d_in[0];
  const float* wqkv  = (const float*)d_in[1];
  const float* wproj = (const float*)d_in[2];
  const float* bproj = (const float*)d_in[3];
  const float* rph   = (const float*)d_in[4];
  const float* rpw   = (const float*)d_in[5];
  float* out = (float*)d_out;

  // workspace carve-up (~38 MB total)
  char* ws = (char*)d_ws;
  const size_t qkv_bytes = (size_t)kBH * kSeq * kHd * sizeof(__bf16);   // 6 MB
  const size_t rel_bytes = (size_t)kBH * kSeq * kHdim * sizeof(float);  // 6 MB
  __bf16* qb    = (__bf16*)ws; ws += qkv_bytes;
  __bf16* kb    = (__bf16*)ws; ws += qkv_bytes;
  __bf16* vb    = (__bf16*)ws; ws += qkv_bytes;
  float*  rel_h = (float*)ws;  ws += rel_bytes;
  float*  rel_w = (float*)ws;  ws += rel_bytes;
  __bf16* ctx   = (__bf16*)ws; ws += qkv_bytes;

  qkv_gemm<<<dim3(kM / 128, kNqkv / 64), 128, 0, stream>>>(x, wqkv, qb, kb, vb);
  relbias<<<dim3((kBH * kSeq * kHdim) / 256), 256, 0, stream>>>(qb, rph, rpw,
                                                                rel_h, rel_w);
  attn_fused<<<dim3(kSeq / 64, kBH), 128, 0, stream>>>(qb, kb, vb, rel_h,
                                                       rel_w, ctx);
  proj_gemm<<<dim3(kM / 128, kC / 64), 128, 0, stream>>>(ctx, wproj, bproj, out);
}
